// TimeSeriesSelfAttention_42898133353109
// MI455X (gfx1250) — matrix operand order repaired and checked by reading, not compiled, not run on hardware
//
#include <hip/hip_runtime.h>
#include <hip/hip_bf16.h>
#include <stdint.h>

// ---------------------------------------------------------------------------
// Problem constants (match reference)
// ---------------------------------------------------------------------------
#define BB   4
#define CH   64
#define NH   8
#define LS   2048
#define HC   (NH * CH)          // 512
#define QT   128                // queries per block (attn)
#define KT   64                 // keys per tile   (attn)
#define NKT  (LS / KT)          // 32 key tiles
#define QK_SCALE 0.3535533906f  // 64^-0.25

typedef __attribute__((ext_vector_type(16))) __bf16 v16bf;
typedef __attribute__((ext_vector_type(8)))  float  v8f;
typedef __attribute__((ext_vector_type(4)))  int    v4i;

union Frag16 {                  // 16 bf16 elements = 8 VGPRs (A or B fragment)
    v16bf    bf;
    uint32_t u[8];
};
union FragC {                   // 16x16 f32 C/D fragment = 8 VGPRs
    v8f   f;
    float x[8];
};

__device__ __forceinline__ uint16_t f32_to_bf16(float f) {
    uint32_t u = __float_as_uint(f);
    u += 0x7FFFu + ((u >> 16) & 1u);      // round-to-nearest-even
    return (uint16_t)(u >> 16);
}
__device__ __forceinline__ float bf16_to_f32(uint16_t h) {
    return __uint_as_float(((uint32_t)h) << 16);
}
// lane <-> lane^16 exchange (SWAPX16): offset = xor(0x10)<<10 | and(0x1f)
__device__ __forceinline__ float swz16(float v) {
    return __int_as_float(__builtin_amdgcn_ds_swizzle(__float_as_int(v), 0x401F));
}

// ---- CDNA5 async global->LDS (ASYNCcnt) with safe fallback ----------------
// Builtin signature (from hipcc diagnostic): param0 = int4 addrspace(1)*,
// param1 = LDS side, then imm offset, imm cpol.
#define AS1_I4(p) ((__attribute__((address_space(1))) v4i*)(p))
#define AS3_I4(p) ((__attribute__((address_space(3))) v4i*)(p))

__device__ __forceinline__ void async_copy_b128(const void* g, void* lds) {
#if __has_builtin(__builtin_amdgcn_global_load_async_to_lds_b128)
    __builtin_amdgcn_global_load_async_to_lds_b128(AS1_I4(g), AS3_I4(lds), 0, 0);
#else
    *(uint4*)lds = *(const uint4*)g;      // sync fallback (VGPR round trip)
#endif
}
__device__ __forceinline__ void wait_async0() {
#if __has_builtin(__builtin_amdgcn_s_wait_asynccnt)
    __builtin_amdgcn_s_wait_asynccnt(0);
#endif
}

// ---------------------------------------------------------------------------
// Stage 1: depthwise(k=3,pad=1) + pointwise(512x64) for q,k,v -> bf16
// grid = B * (L/64), block = 256
// ---------------------------------------------------------------------------
__global__ __launch_bounds__(256) void dsconv_fused(
    const float* __restrict__ x,
    const float* __restrict__ qdw, const float* __restrict__ qdb,
    const float* __restrict__ qpw, const float* __restrict__ qpb,
    const float* __restrict__ kdw, const float* __restrict__ kdb,
    const float* __restrict__ kpw, const float* __restrict__ kpb,
    const float* __restrict__ vdw, const float* __restrict__ vdb,
    const float* __restrict__ vpw, const float* __restrict__ vpb,
    uint16_t* __restrict__ qo, uint16_t* __restrict__ ko, uint16_t* __restrict__ vo)
{
    __shared__ float    xs[CH][66];        // halo tile: l0-1 .. l0+64
    __shared__ uint16_t ys[3][CH][65];     // depthwise results (bf16), odd dword stride

    const int t   = threadIdx.x;
    const int blk = blockIdx.x;
    const int b   = blk >> 5;              // L/64 = 32 tiles
    const int l0  = (blk & 31) * 64;

    for (int idx = t; idx < CH * 66; idx += 256) {
        int c = idx / 66, i = idx % 66;
        int l = l0 - 1 + i;
        xs[c][i] = (l >= 0 && l < LS) ? x[((size_t)b * CH + c) * LS + l] : 0.f;
    }
    __syncthreads();

    #pragma unroll
    for (int s = 0; s < 3; ++s) {
        const float* dw = (s == 0) ? qdw : (s == 1) ? kdw : vdw;
        const float* db = (s == 0) ? qdb : (s == 1) ? kdb : vdb;
        for (int idx = t; idx < CH * 64; idx += 256) {
            int c = idx >> 6, i = idx & 63;
            float y = dw[c * 3 + 0] * xs[c][i]
                    + dw[c * 3 + 1] * xs[c][i + 1]
                    + dw[c * 3 + 2] * xs[c][i + 2] + db[c];
            ys[s][c][i] = f32_to_bf16(y);
        }
    }
    __syncthreads();

    const int l = t & 63;
    #pragma unroll
    for (int s = 0; s < 3; ++s) {
        const float* pw = (s == 0) ? qpw : (s == 1) ? kpw : vpw;
        const float* pb = (s == 0) ? qpb : (s == 1) ? kpb : vpb;
        uint16_t*   dst = (s == 0) ? qo  : (s == 1) ? ko  : vo;
        const float sc  = (s < 2) ? QK_SCALE : 1.0f;
        for (int o = t >> 6; o < HC; o += 4) {
            float acc = pb[o];
            #pragma unroll 8
            for (int c = 0; c < CH; ++c)
                acc += pw[o * CH + c] * bf16_to_f32(ys[s][c][l]);
            dst[((size_t)b * HC + o) * LS + l0 + l] = f32_to_bf16(acc * sc);
        }
    }
}

// ---------------------------------------------------------------------------
// Stage 2: flash attention per (b,h).  S = K^T Q, softmax over keys, O = V P.
// grid = B*H*(L/QT) = 512, block = 256 (8 waves, 16 queries/wave)
// Double-buffered K/V tiles fed by async global->LDS DMA.
// ---------------------------------------------------------------------------
__global__ __launch_bounds__(256) void flash_attn(
    const uint16_t* __restrict__ qg, const uint16_t* __restrict__ kg,
    const uint16_t* __restrict__ vg, float* __restrict__ og)
{
    __shared__ uint16_t QsT[QT][66];       // QsT[j][c]   (66: odd dword stride)
    __shared__ uint16_t Kr[2][CH][72];     // raw K tile [c][key]; 72 halves = 144B rows
    __shared__ uint16_t Vr[2][CH][72];     //   (16B-aligned rows for b128 async DMA)
    __shared__ uint16_t PsT[8][16][66];    // per-wave P^T: PsT[w][q][key]

    const int t    = threadIdx.x;
    const int lane = t & 31;
    const int w    = t >> 5;
    const int qcol = lane & 15;            // N of every fragment this lane touches
    const int hi   = lane >> 4;            // lane half: 0 / 1

    const int blk  = blockIdx.x;
    const int qblk = blk & 15;             // L/QT = 16
    const int bh   = blk >> 4;             // 0..31
    const size_t base = (size_t)bh * CH * LS;
    const int j0 = qblk * QT;

    // ---- issue async DMA for key-tile 0 while staging Q ----
    #pragma unroll
    for (int i = 0; i < 2; ++i) {          // 512 chunks of 16B per tile
        int id = t + 256 * i;
        int c = id >> 3, g = id & 7;
        const size_t goff = base + (size_t)c * LS + g * 8;
        async_copy_b128(kg + goff, &Kr[0][c][g * 8]);
        async_copy_b128(vg + goff, &Vr[0][c][g * 8]);
    }

    // ---- stage Q tile transposed ----
    for (int idx = t; idx < CH * (QT / 8); idx += 256) {
        int c = idx >> 4, jg = idx & 15;
        uint16_t tmp[8];
        *(uint4*)tmp = *(const uint4*)(qg + base + (size_t)c * LS + j0 + jg * 8);
        #pragma unroll
        for (int i = 0; i < 8; ++i) QsT[jg * 8 + i][c] = tmp[i];
    }
    __syncthreads();

    // ---- Q B-fragments (B 32x16: lane N=qcol, kb=hi*8; v<4 -> K=kb+2v else kb+16+2(v-4))
    Frag16 bq0, bq1;
    {
        const int j = w * 16 + qcol, kb = hi * 8;
        #pragma unroll
        for (int v = 0; v < 8; ++v) {
            bq0.u[v] = *(const uint32_t*)&QsT[j][kb + 2 * v + (v < 4 ? 0 : 8)];
            bq1.u[v] = *(const uint32_t*)&QsT[j][32 + kb + 2 * v + (v < 4 ? 0 : 8)];
        }
    }

    float m = -1e30f, ssum = 0.f;
    FragC O[4];                            // c-tiles 0..3 (each 16c x 16q)
    #pragma unroll
    for (int ct = 0; ct < 4; ++ct)
        #pragma unroll
        for (int r = 0; r < 8; ++r) O[ct].x[r] = 0.f;

    for (int kt = 0; kt < NKT; ++kt) {
        const int cur = kt & 1;
        wait_async0();                     // own DMA for buf[cur] landed
        __syncthreads();                   // all waves' DMA landed; old reads done

        if (kt + 1 < NKT) {                // DMA next tile into other buffer
            const int kb1 = (kt + 1) * KT;
            #pragma unroll
            for (int i = 0; i < 2; ++i) {
                int id = t + 256 * i;
                int c = id >> 3, g = id & 7;
                const size_t goff = base + (size_t)c * LS + kb1 + g * 8;
                async_copy_b128(kg + goff, &Kr[cur ^ 1][c][g * 8]);
                async_copy_b128(vg + goff, &Vr[cur ^ 1][c][g * 8]);
            }
        }

        const uint16_t (*Kc)[72] = Kr[cur];
        const uint16_t (*Vc)[72] = Vr[cur];

        // ---- S = K^T Q : 4 key-subtiles x (2 wmma over c) ----
        // A 16x32: lane M=key=sub*16+qcol, kb=hi*8; v<4 -> K=kb+2v else kb+16+2(v-4)
        FragC S[4];
        #pragma unroll
        for (int sub = 0; sub < 4; ++sub) {
            Frag16 a0, a1;
            const int key = sub * 16 + qcol, kb = hi * 8;
            #pragma unroll
            for (int v = 0; v < 8; ++v) {
                int c0 = (v < 4) ? (kb + 2 * v) : (kb + 16 + 2 * (v - 4));
                uint32_t w0 = Kc[c0][key],      w1 = Kc[c0 + 1][key];
                uint32_t w2 = Kc[c0 + 32][key], w3 = Kc[c0 + 33][key];
                a0.u[v] = w0 | (w1 << 16);
                a1.u[v] = w2 | (w3 << 16);
            }
            v8f acc = {};
            acc = __builtin_amdgcn_wmma_f32_16x16x32_bf16(false, a0.bf, false, bq0.bf,
                                                          (short)0, acc, false, false);
            acc = __builtin_amdgcn_wmma_f32_16x16x32_bf16(false, a1.bf, false, bq1.bf,
                                                          (short)0, acc, false, false);
            S[sub].f = acc;
        }

        // ---- online softmax over 64 keys (rows), per query column ----
        float tmax = -1e30f;
        #pragma unroll
        for (int sub = 0; sub < 4; ++sub)
            #pragma unroll
            for (int r = 0; r < 8; ++r) tmax = fmaxf(tmax, S[sub].x[r]);
        tmax = fmaxf(tmax, swz16(tmax));
        const float mnew = fmaxf(m, tmax);
        const float corr = __expf(m - mnew);
        float psum = 0.f;
        #pragma unroll
        for (int sub = 0; sub < 4; ++sub) {
            #pragma unroll
            for (int r = 0; r < 8; ++r) {
                float p = __expf(S[sub].x[r] - mnew);
                psum += p;
                PsT[w][qcol][sub * 16 + r + hi * 8] = f32_to_bf16(p);
            }
        }
        psum += swz16(psum);
        ssum = ssum * corr + psum;
        m = mnew;
        #pragma unroll
        for (int ct = 0; ct < 4; ++ct)
            #pragma unroll
            for (int r = 0; r < 8; ++r) O[ct].x[r] *= corr;

        // ---- O += V * P ----
        Frag16 bp0, bp1;
        {
            const int kb = hi * 8;
            #pragma unroll
            for (int v = 0; v < 8; ++v) {
                bp0.u[v] = *(const uint32_t*)&PsT[w][qcol][kb + 2 * v + (v < 4 ? 0 : 8)];
                bp1.u[v] = *(const uint32_t*)&PsT[w][qcol][32 + kb + 2 * v + (v < 4 ? 0 : 8)];
            }
        }
        #pragma unroll
        for (int ct = 0; ct < 4; ++ct) {
            Frag16 av0, av1;
            const int c = ct * 16 + qcol, kb = hi * 8;
            #pragma unroll
            for (int v = 0; v < 8; ++v) {
                int k0 = (v < 4) ? (kb + 2 * v) : (kb + 16 + 2 * (v - 4));
                av0.u[v] = *(const uint32_t*)&Vc[c][k0];
                av1.u[v] = *(const uint32_t*)&Vc[c][32 + k0];
            }
            O[ct].f = __builtin_amdgcn_wmma_f32_16x16x32_bf16(false, av0.bf, false, bp0.bf,
                                                              (short)0, O[ct].f, false, false);
            O[ct].f = __builtin_amdgcn_wmma_f32_16x16x32_bf16(false, av1.bf, false, bp1.bf,
                                                              (short)0, O[ct].f, false, false);
        }
    }

    // ---- normalize + store ----
    const float inv = 1.f / ssum;
    #pragma unroll
    for (int ct = 0; ct < 4; ++ct)
        #pragma unroll
        for (int r = 0; r < 8; ++r) {
            int c = ct * 16 + r + hi * 8;          // C/D layout: M = r + 8*hi
            og[base + (size_t)c * LS + j0 + w * 16 + qcol] = O[ct].x[r] * inv;
        }
}

// ---------------------------------------------------------------------------
// Stage 3: unify heads: out[b,c,l] = sum_k u_w[c,k]*att[b,k,l] + u_b[c]
// grid = B*(L/64) = 128, block = 256
// ---------------------------------------------------------------------------
__global__ __launch_bounds__(256) void unify_heads(
    const float* __restrict__ att, const float* __restrict__ uw,
    const float* __restrict__ ub, float* __restrict__ out)
{
    const int t = threadIdx.x, blk = blockIdx.x;
    const int b = blk >> 5, l0 = (blk & 31) * 64;
    const int l = t & 63;
    const float* arow = att + (size_t)b * HC * LS + l0 + l;
    for (int c = t >> 6; c < CH; c += 4) {
        float acc = ub[c];
        const float* wrow = uw + c * HC;
        #pragma unroll 8
        for (int k = 0; k < HC; ++k) acc += wrow[k] * arow[(size_t)k * LS];
        out[((size_t)b * CH + c) * LS + l0 + l] = acc;
    }
}

// ---------------------------------------------------------------------------
extern "C" void kernel_launch(void* const* d_in, const int* in_sizes, int n_in,
                              void* d_out, int out_size, void* d_ws, size_t ws_size,
                              hipStream_t stream) {
    const float* x   = (const float*)d_in[0];
    const float* qdw = (const float*)d_in[1];  const float* qdb = (const float*)d_in[2];
    const float* qpw = (const float*)d_in[3];  const float* qpb = (const float*)d_in[4];
    const float* kdw = (const float*)d_in[5];  const float* kdb = (const float*)d_in[6];
    const float* kpw = (const float*)d_in[7];  const float* kpb = (const float*)d_in[8];
    const float* vdw = (const float*)d_in[9];  const float* vdb = (const float*)d_in[10];
    const float* vpw = (const float*)d_in[11]; const float* vpb = (const float*)d_in[12];
    const float* uw  = (const float*)d_in[13]; const float* ub  = (const float*)d_in[14];

    const size_t nqkv = (size_t)BB * HC * LS;          // 4M elements each
    uint16_t* qws = (uint16_t*)d_ws;
    uint16_t* kws = qws + nqkv;
    uint16_t* vws = kws + nqkv;
    float*    att = (float*)(vws + nqkv);              // 24MB offset, 4B aligned

    dsconv_fused<<<BB * (LS / 64), 256, 0, stream>>>(
        x, qdw, qdb, qpw, qpb, kdw, kdb, kpw, kpb, vdw, vdb, vpw, vpb,
        qws, kws, vws);
    flash_attn<<<BB * NH * (LS / QT), 256, 0, stream>>>(qws, kws, vws, att);
    unify_heads<<<BB * (LS / 64), 256, 0, stream>>>(att, uw, ub, (float*)d_out);
}
